// VectorQuantizer_63264868270490
// MI455X (gfx1250) — matrix operand order repaired and checked by reading, not compiled, not run on hardware
//
#include <hip/hip_runtime.h>
#include <hip/hip_bf16.h>

// CDNA5 / gfx1250: wave32, WMMA 16x16x32 bf16 with f32 accumulate.
typedef __attribute__((ext_vector_type(16))) __bf16 v16bf;
typedef __attribute__((ext_vector_type(8)))  float  v8f;

#define D_DIM          64
#define TILE_K         16          // codes per tile (1296 = 81 * 16)
#define ROWS_PER_WAVE  16
#define WAVES_PER_WG   8
#define ROWS_PER_WG    (ROWS_PER_WAVE * WAVES_PER_WG)  // 128
#define WG_THREADS     256

// ---------------- kernel 0: exact fp32 ||e_k||^2 ----------------
__global__ void vq_enorm_kernel(const float* __restrict__ emb,
                                float* __restrict__ enorm, int K) {
  int k = blockIdx.x * blockDim.x + threadIdx.x;
  if (k >= K) return;
  const float* p = emb + (size_t)k * D_DIM;
  float s = 0.f;
#pragma unroll
  for (int d = 0; d < D_DIM; ++d) { float v = p[d]; s = fmaf(v, v, s); }
  enorm[k] = s;
}

// ---------------- kernel 1: WMMA argmin over codes ----------------
__global__ __launch_bounds__(WG_THREADS)
void vq_argmin_kernel(const float* __restrict__ X, const float* __restrict__ E,
                      const float* __restrict__ Enorm, int* __restrict__ codes,
                      int K) {
  // double-buffered bf16 hi/lo embedding tile: [2][16 codes][64 d], d contiguous
  __shared__ __align__(32) __bf16 sH[2][TILE_K * D_DIM];
  __shared__ __align__(32) __bf16 sL[2][TILE_K * D_DIM];

  const int tid  = threadIdx.x;
  const int wave = tid >> 5;      // wave32 on gfx1250
  const int lane = tid & 31;
  const int rowbase = (blockIdx.x * WAVES_PER_WG + wave) * ROWS_PER_WAVE;

  // ---- build A fragments (16 rows x 64 K) as bf16 hi/lo, contiguous k order ----
  // lanes 0-15: elems 0..15 -> K kb..kb+15 (kb=0)
  // lanes 16-31: kb = 16
  const int kb = (lane < 16) ? 0 : 16;
  const float* rp = X + (size_t)(rowbase + (lane & 15)) * D_DIM;
  v16bf a0h, a0l, a1h, a1l;   // K 0..31 and K 32..63
#pragma unroll
  for (int e = 0; e < 8; ++e) {
    float f;
    f = rp[kb + e];      { __bf16 h = (__bf16)f; a0h[e]     = h; a0l[e]     = (__bf16)(f - (float)h); }
    f = rp[kb + 8 + e]; { __bf16 h = (__bf16)f; a0h[8 + e] = h; a0l[8 + e] = (__bf16)(f - (float)h); }
    f = rp[kb + 32 + e]; { __bf16 h = (__bf16)f; a1h[e]     = h; a1l[e]     = (__bf16)(f - (float)h); }
    f = rp[kb + 40 + e]; { __bf16 h = (__bf16)f; a1h[8 + e] = h; a1l[8 + e] = (__bf16)(f - (float)h); }
  }

  float bestS[8];
  int   bestI[8];
#pragma unroll
  for (int r = 0; r < 8; ++r) { bestS[r] = 3.4e38f; bestI[r] = 0; }

  const int ntiles = K / TILE_K;          // 81
  const int cod  = lane & 15;             // B column / D column
  const int half = (lane >> 4) << 4;      // B K-offset per lane half

  // software pipeline: prefetch tile 0
  float4 vstage = ((const float4*)E)[tid];

  for (int t = 0; t < ntiles; ++t) {
    const int base = t * TILE_K;
    const int buf  = t & 1;

    // stage tile t: fp32 -> bf16 hi/lo into LDS buffer (4 elems/thread)
    {
      float fv[4] = {vstage.x, vstage.y, vstage.z, vstage.w};
#pragma unroll
      for (int j = 0; j < 4; ++j) {
        __bf16 h = (__bf16)fv[j];
        sH[buf][tid * 4 + j] = h;
        sL[buf][tid * 4 + j] = (__bf16)(fv[j] - (float)h);
      }
    }
    __syncthreads();   // single barrier/iter: stage(t) visible; also proves
                       // compute(t-2) readers of buf[t&1] have finished.
                       // Only dscnt needs draining here — prefetch comes after.

    // prefetch tile t+1 AFTER the barrier so its latency overlaps the WMMAs
    {
      const int tn = (t + 1 < ntiles) ? (t + 1) : t;
      vstage = ((const float4*)(E + (size_t)tn * TILE_K * D_DIM))[tid];
    }

    // B fragments (32x16): lane = column, element e -> K = half + e (+32 for b1)
    const v16bf b0h = *(const v16bf*)&sH[buf][cod * D_DIM + half];
    const v16bf b1h = *(const v16bf*)&sH[buf][cod * D_DIM + 32 + half];
    const v16bf b0l = *(const v16bf*)&sL[buf][cod * D_DIM + half];
    const v16bf b1l = *(const v16bf*)&sL[buf][cod * D_DIM + 32 + half];

    // split-bf16 dot: hi*hi + hi*lo + lo*hi  (~fp32 fidelity), K=64 total
    v8f acc = {};
    acc = __builtin_amdgcn_wmma_f32_16x16x32_bf16(false, a0h, false, b0h, (short)0, acc, false, false);
    acc = __builtin_amdgcn_wmma_f32_16x16x32_bf16(false, a1h, false, b1h, (short)0, acc, false, false);
    acc = __builtin_amdgcn_wmma_f32_16x16x32_bf16(false, a0h, false, b0l, (short)0, acc, false, false);
    acc = __builtin_amdgcn_wmma_f32_16x16x32_bf16(false, a1h, false, b1l, (short)0, acc, false, false);
    acc = __builtin_amdgcn_wmma_f32_16x16x32_bf16(false, a0l, false, b0h, (short)0, acc, false, false);
    acc = __builtin_amdgcn_wmma_f32_16x16x32_bf16(false, a1l, false, b1h, (short)0, acc, false, false);

    const float en  = Enorm[base + cod];  // ||e||^2, exact fp32 (L2-hot)
    const int   idx = base + cod;
#pragma unroll
    for (int r = 0; r < 8; ++r) {         // D row = r + 8*(lane/16)
      float s = en - 2.0f * acc[r];
      if (s < bestS[r]) { bestS[r] = s; bestI[r] = idx; }  // strict < : first-index tie-break
    }
  }

  // reduce across the 16 columns held by each 16-lane half
#pragma unroll
  for (int m = 1; m < 16; m <<= 1) {
#pragma unroll
    for (int r = 0; r < 8; ++r) {
      float os = __shfl_xor(bestS[r], m, 16);
      int   oi = __shfl_xor(bestI[r], m, 16);
      if (os < bestS[r] || (os == bestS[r] && oi < bestI[r])) { bestS[r] = os; bestI[r] = oi; }
    }
  }

  if ((lane & 15) == 0) {
    const int mbase = rowbase + ((lane >> 4) << 3);   // lane 0 -> rows 0..7, lane 16 -> rows 8..15
#pragma unroll
    for (int r = 0; r < 8; ++r) codes[mbase + r] = bestI[r];
  }
}

// ---------------- kernel 2: gather code vectors (HBM-bound) ----------------
__global__ void vq_gather_kernel(const int* __restrict__ codes,
                                 const float* __restrict__ E,
                                 float* __restrict__ out, int N) {
  int t = blockIdx.x * blockDim.x + threadIdx.x;
  if (t >= N * 16) return;
  int row = t >> 4, part = t & 15;      // 16 float4 per 64-float row
  int c = codes[row];
  float4 v = ((const float4*)E)[(size_t)c * 16 + part];
  ((float4*)out)[(size_t)row * 16 + part] = v;
}

extern "C" void kernel_launch(void* const* d_in, const int* in_sizes, int n_in,
                              void* d_out, int out_size, void* d_ws, size_t ws_size,
                              hipStream_t stream) {
  const float* X = (const float*)d_in[0];        // [32,64,64,64] fp32
  const float* E = (const float*)d_in[1];        // [1296,64] fp32
  const int N = in_sizes[0] / D_DIM;             // 131072 points
  const int K = in_sizes[1] / D_DIM;             // 1296 codes

  float* enorm = (float*)d_ws;                   // 1296 floats of scratch
  int*   codes = (int*)d_out;                    // output 0: int32 codes [N]
  float* vecs  = (float*)d_out + N;              // output 1: fp32 code_vecs [N,64]

  vq_enorm_kernel<<<(K + 255) / 256, 256, 0, stream>>>(E, enorm, K);
  vq_argmin_kernel<<<N / ROWS_PER_WG, WG_THREADS, 0, stream>>>(X, E, enorm, codes, K);
  vq_gather_kernel<<<(N * 16 + 255) / 256, 256, 0, stream>>>(codes, E, vecs, N);
}
